// FeatureFusion_63462436766085
// MI455X (gfx1250) — matrix operand order repaired and checked by reading, not compiled, not run on hardware
//
#include <hip/hip_runtime.h>
#include <stdint.h>

// ---------------------------------------------------------------------------
// Types for CDNA5 WMMA (gfx1250, wave32)
// ---------------------------------------------------------------------------
typedef __attribute__((ext_vector_type(16))) __bf16 v16bf;
typedef __attribute__((ext_vector_type(8)))  float  v8f;

#define DM   256          // d_model
#define ROWS 25600        // 256 seqs * 100 tokens
#define ATTN_LDS 108544   // qs(7168)+ks(7168)+vts(8192)+S(57344)+P(28672)

__device__ __forceinline__ unsigned short f2bf(float x) {
  unsigned int u = __float_as_uint(x);
  u = (u + 0x7fffu + ((u >> 16) & 1u)) >> 16;   // round-to-nearest-even
  return (unsigned short)u;
}
__device__ __forceinline__ float bf2f(unsigned short b) {
  return __uint_as_float(((unsigned int)b) << 16);
}
__device__ __forceinline__ v16bf frag16(const unsigned short* p0, const unsigned short* p1) {
  v16bf r;
  uint4* u = reinterpret_cast<uint4*>(&r);
  u[0] = *reinterpret_cast<const uint4*>(p0);
  u[1] = *reinterpret_cast<const uint4*>(p1);
  return r;
}
__device__ __forceinline__ v8f wmma_bf16(v16bf a, v16bf b, v8f c) {
  return __builtin_amdgcn_wmma_f32_16x16x32_bf16(false, a, false, b, (short)0, c, false, false);
}

// CDNA5 async global->LDS DMA (16B per lane), tracked by ASYNCcnt.
__device__ __forceinline__ void async_ld16(void* lds, const void* gaddr) {
  asm volatile("global_load_async_to_lds_b128 %0, %1, off"
               :: "v"((unsigned)(uintptr_t)lds),
                  "v"((unsigned long long)(uintptr_t)gaddr)
               : "memory");
}
__device__ __forceinline__ void wait_async(int pending) {
  if (pending) asm volatile("s_wait_asynccnt 0x3" ::: "memory");
  else         asm volatile("s_wait_asynccnt 0x0" ::: "memory");
}

// ---------------------------------------------------------------------------
// Elementwise helpers
// ---------------------------------------------------------------------------
__global__ void k_cvt(const float* __restrict__ s, unsigned short* __restrict__ d, size_t n) {
  size_t i = (size_t)blockIdx.x * blockDim.x + threadIdx.x;
  if (i < n) d[i] = f2bf(s[i]);
}
__global__ void k_cvt_relu(const float* __restrict__ s, unsigned short* __restrict__ d, size_t n) {
  size_t i = (size_t)blockIdx.x * blockDim.x + threadIdx.x;
  if (i < n) d[i] = f2bf(fmaxf(s[i], 0.0f));
}
// bf16 transpose-convert for weights: d[c*R + r] = bf16(s[r*C + c])
__global__ void k_cvt_t(const float* __restrict__ s, unsigned short* __restrict__ d, int R, int C) {
  size_t i = (size_t)blockIdx.x * blockDim.x + threadIdx.x;
  if (i >= (size_t)R * C) return;
  int r = (int)(i / C), c = (int)(i % C);
  d[(size_t)c * R + r] = f2bf(s[i]);
}

// h[r, 0:256] = x[r], h[r, 256:512] = msg[bcast ? r/100 : r]  (bf16 out)
__global__ void k_concat(const float* __restrict__ x, const float* __restrict__ msg,
                         unsigned short* __restrict__ h, int bcast) {
  size_t i = (size_t)blockIdx.x * blockDim.x + threadIdx.x;
  if (i >= (size_t)ROWS * 512) return;
  int r = (int)(i >> 9);
  int c = (int)(i & 511);
  float v = (c < 256) ? x[(size_t)r * 256 + c]
                      : msg[(size_t)(bcast ? (r / 100) : r) * 256 + (c - 256)];
  h[i] = f2bf(v);
}

// LayerNorm over last dim (256); optional residual add. One block per row.
__global__ void __launch_bounds__(256)
k_ln(const float* __restrict__ in, const float* __restrict__ g, const float* __restrict__ bb,
     float* __restrict__ out, const float* __restrict__ residual, int hasRes) {
  __shared__ float red[256];
  int row = blockIdx.x, c = threadIdx.x;
  size_t idx = (size_t)row * 256 + c;
  float x = in[idx];
  red[c] = x; __syncthreads();
  for (int s = 128; s > 0; s >>= 1) { if (c < s) red[c] += red[c + s]; __syncthreads(); }
  float mean = red[0] * (1.0f / 256.0f);
  __syncthreads();
  float d = x - mean;
  red[c] = d * d; __syncthreads();
  for (int s = 128; s > 0; s >>= 1) { if (c < s) red[c] += red[c + s]; __syncthreads(); }
  float var = red[0] * (1.0f / 256.0f);
  float y = d * rsqrtf(var + 1e-5f) * g[c] + bb[c];
  out[idx] = hasRes ? (residual[idx] + y) : y;
}

// s2s shortcut: sum_w softmax == 1  =>  msg[m,:,c] = sum_e v[m,e,c]
__global__ void k_vsum(const float* __restrict__ v, float* __restrict__ out) {
  int m = blockIdx.x, c = threadIdx.x;
  const float* p = v + (size_t)m * 100 * 256 + c;
  float s = 0.f;
  for (int e = 0; e < 100; ++e) s += p[(size_t)e * 256];
  out[(size_t)m * 256 + c] = s;
}

// F.normalize(dim=1) over the 400 rows of each group b; writes bf16.
__global__ void k_qnorm(const float* __restrict__ qf, unsigned short* __restrict__ qbf, int R) {
  int col = blockIdx.x * 256 + threadIdx.x;   // 64*256 columns
  int b = col >> 8, c = col & 255;
  const float* base = qf + (size_t)b * R * 256 + c;
  unsigned short* ob = qbf + (size_t)b * R * 256 + c;
  float s = 0.f;
  for (int r = 0; r < R; ++r) { float x = base[(size_t)r * 256]; s += x * x; }
  float inv = 1.f / fmaxf(sqrtf(s), 1e-12f);
  for (int r = 0; r < R; ++r) ob[(size_t)r * 256] = f2bf(base[(size_t)r * 256] * inv);
}

// ---------------------------------------------------------------------------
// bf16 WMMA GEMM: C[MxN] = A[MxK] * B[KxN] with B pre-transposed (BT = [N][K]).
// Block tile 128x64, 8 waves; double-buffered LDS filled by async DMA so the
// next tile's global->LDS transfer overlaps the current tile's WMMAs.
// M%128==0, N%64==0, K%32==0 (all call sites satisfy this).
// ---------------------------------------------------------------------------
__global__ void __launch_bounds__(256)
k_gemm(const unsigned short* __restrict__ A, const unsigned short* __restrict__ BT,
       float* __restrict__ C, int M, int N, int K, int lda, int ldbt, int ldc) {
  __shared__ unsigned short As[2][128 * 32];   // row-major M x K tile
  __shared__ unsigned short Bs[2][64 * 32];    // [n][k]
  int tid = threadIdx.x, lane = tid & 31, wv = tid >> 5;
  int lo = lane & 15, hi = lane >> 4;
  int m0 = blockIdx.y * 128, n0 = blockIdx.x * 64;
  int wrow = wv * 16;
  v8f acc[4] = {};

  auto issue = [&](int buf, int k0) {         // 3 async instrs per wave
    for (int i = tid; i < 512; i += 256) {    // A: 128x32 in 8-elem chunks
      int r = i >> 2, c = (i & 3) * 8;
      async_ld16(&As[buf][r * 32 + c], &A[(size_t)(m0 + r) * lda + k0 + c]);
    }
    {                                         // BT: 64x32 in 8-elem chunks
      int n = tid >> 2, c = (tid & 3) * 8;
      async_ld16(&Bs[buf][n * 32 + c], &BT[(size_t)(n0 + n) * ldbt + k0 + c]);
    }
  };

  int nt = K >> 5;
  issue(0, 0);
  for (int t = 0; t < nt; ++t) {
    int cur = t & 1;
    int hasNext = (t + 1 < nt);
    if (hasNext) issue(cur ^ 1, (t + 1) << 5);
    wait_async(hasNext);                      // current tile complete (in-order)
    __syncthreads();
    const unsigned short* ap = &As[cur][(wrow + lo) * 32 + hi * 8];
    v16bf a = frag16(ap, ap + 16);
    const unsigned short* bp0 = &Bs[cur][(0 * 16 + lo) * 32 + hi * 8];
    const unsigned short* bp1 = &Bs[cur][(1 * 16 + lo) * 32 + hi * 8];
    const unsigned short* bp2 = &Bs[cur][(2 * 16 + lo) * 32 + hi * 8];
    const unsigned short* bp3 = &Bs[cur][(3 * 16 + lo) * 32 + hi * 8];
    v16bf b0 = frag16(bp0, bp0 + 16);
    v16bf b1 = frag16(bp1, bp1 + 16);
    v16bf b2 = frag16(bp2, bp2 + 16);
    v16bf b3 = frag16(bp3, bp3 + 16);
    acc[0] = wmma_bf16(a, b0, acc[0]);
    acc[1] = wmma_bf16(a, b1, acc[1]);
    acc[2] = wmma_bf16(a, b2, acc[2]);
    acc[3] = wmma_bf16(a, b3, acc[3]);
    __syncthreads();
  }
#pragma unroll
  for (int t = 0; t < 4; ++t)
#pragma unroll
    for (int j = 0; j < 8; ++j)
      C[(size_t)(m0 + wrow + j + hi * 8) * ldc + n0 + t * 16 + lo] = acc[t][j];
}

// ---------------------------------------------------------------------------
// s2m fused windowed attention, phase 1.
// One block per (b, h, w): S = q(400x32) @ k_w^T(32x100), softmax over 100,
// qv = P @ v_w (100x32). q processed in 4 chunks of 100 rows (tiles padded to
// 112, PV K padded to 128). q/k tiles arrive via async global->LDS DMA.
// ---------------------------------------------------------------------------
__global__ void __launch_bounds__(256)
k_attn(const unsigned short* __restrict__ qbf, const unsigned short* __restrict__ kbf,
       const unsigned short* __restrict__ vbf, unsigned short* __restrict__ qvbf) {
  extern __shared__ char smem[];
  unsigned short* qs  = (unsigned short*)smem;                     // 112 x 32
  unsigned short* ks  = qs + 112 * 32;                             // 112 x 32
  unsigned short* vts = ks + 112 * 32;                             // 32 x 128 (v^T)
  float*          S   = (float*)(smem + 22528);                    // 112 x 128
  unsigned short* P   = (unsigned short*)(smem + 22528 + 112 * 128 * 4); // 112x128

  const int tid = threadIdx.x, lane = tid & 31, wv = tid >> 5;
  const int lo = lane & 15, hi = lane >> 4;
  const int blk = blockIdx.x;
  const int w = blk & 3, h = (blk >> 2) & 7, b = blk >> 5;
  const float temp = 0.17677669529663687f;                         // 1/sqrt(32)

  size_t kvbase = ((size_t)(b * 400 + w * 100)) * 256 + h * 32;

  // zero the pad regions (disjoint from all fills below; no barrier needed)
  for (int i = tid; i < 12 * 32; i += 256) {            // qs/ks rows 100..111
    qs[100 * 32 + i] = 0;
    ks[100 * 32 + i] = 0;
  }
  for (int i = tid; i < 32 * 28; i += 256)              // vts cols 100..127
    vts[(i / 28) * 128 + 100 + (i % 28)] = 0;

  // async DMA the 100 valid k rows (400 x 16B chunks)
  for (int i = tid; i < 400; i += 256) {
    int r = i >> 2, c = (i & 3) * 8;
    async_ld16(&ks[r * 32 + c], &kbf[kvbase + (size_t)r * 256 + c]);
  }
  // v^T: vectorized global reads, transposed ds stores
  for (int i = tid; i < 400; i += 256) {
    int r = i >> 2, c = (i & 3) * 8;
    uint4 d4 = *reinterpret_cast<const uint4*>(&vbf[kvbase + (size_t)r * 256 + c]);
    const unsigned short* e = reinterpret_cast<const unsigned short*>(&d4);
#pragma unroll
    for (int j = 0; j < 8; ++j) vts[(c + j) * 128 + r] = e[j];
  }

  for (int qc = 0; qc < 4; ++qc) {
    size_t qbase = ((size_t)(b * 400 + qc * 100)) * 256 + h * 32;
    for (int i = tid; i < 400; i += 256) {
      int r = i >> 2, c = (i & 3) * 8;
      async_ld16(&qs[r * 32 + c], &qbf[qbase + (size_t)r * 256 + c]);
    }
    wait_async(0);
    __syncthreads();

    // --- S = q @ k^T : 7x7 tiles, single K step (K=32) ---
    for (int job = wv; job < 49; job += 8) {
      int m0 = (job / 7) * 16, n0 = (job % 7) * 16;
      const unsigned short* ap = &qs[(m0 + lo) * 32 + hi * 8];
      v16bf a = frag16(ap, ap + 16);
      const unsigned short* bp = &ks[(n0 + lo) * 32 + hi * 8];
      v16bf bm = frag16(bp, bp + 16);
      v8f c = {};
      c = wmma_bf16(a, bm, c);
#pragma unroll
      for (int j = 0; j < 8; ++j) S[(m0 + j + hi * 8) * 128 + n0 + lo] = c[j];
    }
    __syncthreads();

    // --- softmax over the 100 valid columns, write bf16 P (padded zero) ---
    if (tid < 112) {
      float* srow = &S[tid * 128];
      unsigned short* prow = &P[tid * 128];
      if (tid < 100) {
        float mx = -1e30f;
        for (int j = 0; j < 100; ++j) { float v = srow[j] * temp; srow[j] = v; if (v > mx) mx = v; }
        float s = 0.f;
        for (int j = 0; j < 100; ++j) { float e = __expf(srow[j] - mx); srow[j] = e; s += e; }
        float inv = 1.f / s;
        for (int j = 0; j < 100; ++j) prow[j] = f2bf(srow[j] * inv);
        for (int j = 100; j < 128; ++j) prow[j] = 0;
      } else {
        for (int j = 0; j < 128; ++j) prow[j] = 0;
      }
    }
    __syncthreads();

    // --- qv = P @ v : 7x2 tiles, K = 128 (4 steps) ---
    for (int job = wv; job < 14; job += 8) {
      int m0 = (job >> 1) * 16, n0 = (job & 1) * 16;
      v8f c = {};
#pragma unroll
      for (int kb = 0; kb < 128; kb += 32) {
        const unsigned short* ap = &P[(m0 + lo) * 128 + kb + hi * 8];
        v16bf a = frag16(ap, ap + 16);
        const unsigned short* bp = &vts[(n0 + lo) * 128 + kb + hi * 8];
        v16bf bm = frag16(bp, bp + 16);
        c = wmma_bf16(a, bm, c);
      }
#pragma unroll
      for (int j = 0; j < 8; ++j) {
        int row = m0 + j + hi * 8;
        if (row < 100) {
          int qg = qc * 100 + row;
          qvbf[(((size_t)(b * 400 + qg) * 4 + w) * 256) + h * 32 + n0 + lo] = f2bf(c[j]);
        }
      }
    }
    __syncthreads();
  }
}

// ---------------------------------------------------------------------------
// s2m phase 2: per (b,q,h) wave — k_in/v_in = qv @ w{k,v}_in (32x32),
// 4-way softmax on q . k_in, msg = sum_w a_w * v_in_w.
// ---------------------------------------------------------------------------
__global__ void __launch_bounds__(256)
k_phase2(const unsigned short* __restrict__ qbf, const unsigned short* __restrict__ qvbf,
         const unsigned short* __restrict__ wkin, const unsigned short* __restrict__ wvin,
         float* __restrict__ msg) {
  __shared__ float wks[32][32];
  __shared__ float wvs[32][32];
  int tid = threadIdx.x;
  for (int i = tid; i < 1024; i += 256) {
    wks[i >> 5][i & 31] = bf2f(wkin[i]);
    wvs[i >> 5][i & 31] = bf2f(wvin[i]);
  }
  __syncthreads();
  int lane = tid & 31;
  int wid = blockIdx.x * 8 + (tid >> 5);    // (b*400+q)*8 + h
  int h = wid & 7, row = wid >> 3;
  float qd = bf2f(qbf[(size_t)row * 256 + h * 32 + lane]);
  float r[4], kin[4], vin[4], lg[4];
#pragma unroll
  for (int w = 0; w < 4; ++w)
    r[w] = bf2f(qvbf[((size_t)row * 4 + w) * 256 + h * 32 + lane]);
#pragma unroll
  for (int w = 0; w < 4; ++w) {
    float kk = 0.f, vv = 0.f;
    for (int j = 0; j < 32; ++j) {
      float a = __shfl(r[w], j, 32);
      kk += a * wks[j][lane];
      vv += a * wvs[j][lane];
    }
    kin[w] = kk; vin[w] = vv;
  }
#pragma unroll
  for (int w = 0; w < 4; ++w) {
    float d = qd * kin[w];
    for (int off = 16; off > 0; off >>= 1) d += __shfl_xor(d, off, 32);
    lg[w] = d * 0.17677669529663687f;
  }
  float mx = fmaxf(fmaxf(lg[0], lg[1]), fmaxf(lg[2], lg[3]));
  float e[4], s = 0.f;
#pragma unroll
  for (int w = 0; w < 4; ++w) { e[w] = __expf(lg[w] - mx); s += e[w]; }
  float inv = 1.f / s, o = 0.f;
#pragma unroll
  for (int w = 0; w < 4; ++w) o += e[w] * inv * vin[w];
  msg[(size_t)row * 256 + h * 32 + lane] = o;
}

// ---------------------------------------------------------------------------
// Host orchestration
// ---------------------------------------------------------------------------
extern "C" void kernel_launch(void* const* d_in, const int* in_sizes, int n_in,
                              void* d_out, int out_size, void* d_ws, size_t ws_size,
                              hipStream_t stream) {
  (void)in_sizes; (void)n_in; (void)out_size; (void)ws_size;
  hipFuncSetAttribute(reinterpret_cast<const void*>(k_attn),
                      hipFuncAttributeMaxDynamicSharedMemorySize, ATTN_LDS);

  size_t off = 0;
  auto alloc = [&](size_t bytes) -> void* {
    off = (off + 255) & ~(size_t)255;
    void* p = (char*)d_ws + off;
    off += bytes;
    return p;
  };
  const size_t NF = (size_t)ROWS * DM;          // 6,553,600

  float*          F0    = (float*)alloc(NF * 4);
  float*          F1    = (float*)alloc(NF * 4);
  unsigned short* XBF   = (unsigned short*)alloc(NF * 2);
  unsigned short* YBF   = (unsigned short*)alloc(NF * 2);
  float*          QF    = (float*)alloc(NF * 4);
  float*          KF    = (float*)alloc(NF * 4);
  float*          VF    = (float*)alloc(NF * 4);
  unsigned short* QBF   = (unsigned short*)alloc(NF * 2);
  unsigned short* KBF   = (unsigned short*)alloc(NF * 2);
  unsigned short* VBF   = (unsigned short*)alloc(NF * 2);
  unsigned short* QVBF  = (unsigned short*)alloc(NF * 4 * 2);     // (25600,4,256) bf16
  float*          MSGF  = (float*)alloc(NF * 4);
  unsigned short* MSGBF = (unsigned short*)alloc(NF * 2);
  float*          MMF   = (float*)alloc(NF * 4);
  unsigned short* HBF   = (unsigned short*)alloc((size_t)ROWS * 512 * 2);
  float*          MIDF  = (float*)alloc((size_t)ROWS * 512 * 4);
  unsigned short* MIDBF = (unsigned short*)alloc((size_t)ROWS * 512 * 2);
  float*          OUT2F = (float*)alloc(NF * 4);
  float*          SM0   = (float*)alloc((size_t)256 * 256 * 4);
  float*          SM1   = (float*)alloc((size_t)256 * 256 * 4);
  unsigned short* SMBF  = (unsigned short*)alloc((size_t)256 * 256 * 2);

  auto cvt = [&](const float* s, unsigned short* d, size_t n) {
    k_cvt<<<(unsigned)((n + 255) / 256), 256, 0, stream>>>(s, d, n);
  };
  // transpose-convert a weight (R x C) to bf16 [C][R]
  auto cvtwT = [&](const float* s, int R, int C) -> unsigned short* {
    size_t n = (size_t)R * C;
    unsigned short* d = (unsigned short*)alloc(n * 2);
    k_cvt_t<<<(unsigned)((n + 255) / 256), 256, 0, stream>>>(s, d, R, C);
    return d;
  };
  auto cvtw = [&](const float* s, size_t n) -> unsigned short* {
    unsigned short* d = (unsigned short*)alloc(n * 2);
    cvt(s, d, n);
    return d;
  };
  // C = A(MxK, bf16) * B(KxN) with BT = bf16 [N][K]
  auto gemm = [&](const unsigned short* A, const unsigned short* BT, float* C,
                  int M, int N, int K) {
    dim3 g(N / 64, M / 128);
    k_gemm<<<g, 256, 0, stream>>>(A, BT, C, M, N, K, K, K, N);
  };

  // ---- parse params (pytree order: feat0, feat1, then sorted-key blocks) ----
  struct WBf { const float *g1, *b1, *g2, *b2;
               unsigned short *wq, *wk, *wv, *wm, *w1, *w2, *wki, *wvi; };
  WBf sa[2], sx[4];
  int idx = 2;
  for (int i = 0; i < 2; ++i) {   // keys: b1,b2,g1,g2,w1,w2,wk,wm,wq,wv
    sa[i].b1 = (const float*)d_in[idx + 0];
    sa[i].b2 = (const float*)d_in[idx + 1];
    sa[i].g1 = (const float*)d_in[idx + 2];
    sa[i].g2 = (const float*)d_in[idx + 3];
    sa[i].w1 = cvtwT((const float*)d_in[idx + 4], 512, 512);
    sa[i].w2 = cvtwT((const float*)d_in[idx + 5], 512, 256);
    sa[i].wk = nullptr;                           // wq/wk provably unused (s2s shortcut)
    sa[i].wm = cvtwT((const float*)d_in[idx + 7], 256, 256);
    sa[i].wq = nullptr;
    sa[i].wv = cvtwT((const float*)d_in[idx + 9], 256, 256);
    sa[i].wki = sa[i].wvi = nullptr;
    idx += 10;
  }
  for (int i = 0; i < 4; ++i) {   // keys: b1,b2,g1,g2,w1,w2,wk,wk_in,wm,wq,wv,wv_in
    sx[i].b1 = (const float*)d_in[idx + 0];
    sx[i].b2 = (const float*)d_in[idx + 1];
    sx[i].g1 = (const float*)d_in[idx + 2];
    sx[i].g2 = (const float*)d_in[idx + 3];
    sx[i].w1 = cvtwT((const float*)d_in[idx + 4], 512, 512);
    sx[i].w2 = cvtwT((const float*)d_in[idx + 5], 512, 256);
    sx[i].wk  = cvtwT((const float*)d_in[idx + 6], 256, 256);
    sx[i].wki = cvtw((const float*)d_in[idx + 7], 32 * 32);   // used row-major in phase2
    sx[i].wm  = cvtwT((const float*)d_in[idx + 8], 256, 256);
    sx[i].wq  = cvtwT((const float*)d_in[idx + 9], 256, 256);
    sx[i].wv  = cvtwT((const float*)d_in[idx + 10], 256, 256);
    sx[i].wvi = cvtw((const float*)d_in[idx + 11], 32 * 32);
    idx += 12;
  }

  hipMemcpyAsync(F0, d_in[0], NF * 4, hipMemcpyDeviceToDevice, stream);
  hipMemcpyAsync(F1, d_in[1], NF * 4, hipMemcpyDeviceToDevice, stream);

  auto mlp_tail = [&](const WBf& p, float* X, const float* msgln, int bcast) {
    k_concat<<<(unsigned)((size_t)ROWS * 512 / 256), 256, 0, stream>>>(X, msgln, HBF, bcast);
    gemm(HBF, p.w1, MIDF, ROWS, 512, 512);
    k_cvt_relu<<<(unsigned)((size_t)ROWS * 512 / 256), 256, 0, stream>>>(MIDF, MIDBF, (size_t)ROWS * 512);
    gemm(MIDBF, p.w2, OUT2F, ROWS, 256, 512);
    k_ln<<<ROWS, 256, 0, stream>>>(OUT2F, p.g2, p.b2, X, X, 1);   // X = X + LN(.)
  };

  auto run_sa = [&](const WBf& p, float* X) {
    cvt(X, XBF, NF);
    gemm(XBF, p.wv, VF, ROWS, 256, 256);
    k_vsum<<<256, 256, 0, stream>>>(VF, SM0);          // msg == sum_e v (softmax sums to 1)
    cvt(SM0, SMBF, (size_t)256 * 256);
    gemm(SMBF, p.wm, SM1, 256, 256, 256);
    k_ln<<<256, 256, 0, stream>>>(SM1, p.g1, p.b1, SM1, SM1, 0);
    mlp_tail(p, X, SM1, /*bcast=*/1);
  };

  auto run_s2m = [&](const WBf& p, float* X, const float* Y) {
    cvt(X, XBF, NF);
    const unsigned short* YB = XBF;
    if (Y != X) { cvt(Y, YBF, NF); YB = YBF; }
    gemm(XBF, p.wq, QF, ROWS, 256, 256);
    gemm(YB,  p.wk, KF, ROWS, 256, 256);
    gemm(YB,  p.wv, VF, ROWS, 256, 256);
    k_qnorm<<<64, 256, 0, stream>>>(QF, QBF, 400);
    cvt(KF, KBF, NF);
    cvt(VF, VBF, NF);
    k_attn<<<64 * 8 * 4, 256, ATTN_LDS, stream>>>(QBF, KBF, VBF, QVBF);
    k_phase2<<<ROWS, 256, 0, stream>>>(QBF, QVBF, p.wki, p.wvi, MSGF);
    cvt(MSGF, MSGBF, NF);
    gemm(MSGBF, p.wm, MMF, ROWS, 256, 256);
    k_ln<<<ROWS, 256, 0, stream>>>(MMF, p.g1, p.b1, MMF, MMF, 0);
    mlp_tail(p, X, MMF, /*bcast=*/0);
  };

  // ---- NAMES = [Sa, SCa, MCa, Sa, SCa, MCa] ----
  run_sa(sa[0], F0);  run_sa(sa[0], F1);
  run_s2m(sx[0], F1, F1);
  run_s2m(sx[1], F0, F1);  run_s2m(sx[1], F1, F0);
  run_sa(sa[1], F0);  run_sa(sa[1], F1);
  run_s2m(sx[2], F1, F1);
  run_s2m(sx[3], F0, F1);  run_s2m(sx[3], F1, F0);

  float* out = (float*)d_out;
  hipMemcpyAsync(out,      F0, NF * 4, hipMemcpyDeviceToDevice, stream);
  hipMemcpyAsync(out + NF, F1, NF * 4, hipMemcpyDeviceToDevice, stream);
}
